// SuperMegnetNodeModel_47974784696399
// MI455X (gfx1250) — matrix operand order repaired and checked by reading, not compiled, not run on hardware
//
#include <hip/hip_runtime.h>
#include <hip/hip_bf16.h>
#include <math.h>

#define N_NODES 50000
#define N_EDGES 800000
#define HDIM    128
#define GDIMF   64
#define NGRP    10
#define LAMDA   0.01f
#define EPSV    1e-5f

typedef __attribute__((ext_vector_type(16))) __bf16 v16bf;
typedef __attribute__((ext_vector_type(8)))  __bf16 v8bf;
typedef __attribute__((ext_vector_type(8)))  float  v8f;
typedef __attribute__((ext_vector_type(4)))  unsigned int u32x4;
typedef __attribute__((ext_vector_type(8)))  int i32x8;
typedef __attribute__((ext_vector_type(4)))  int i32x4;

#if defined(__has_builtin)
#if __has_builtin(__builtin_amdgcn_tensor_load_to_lds)
#define HAVE_TDM 1
#endif
#if __has_builtin(__builtin_amdgcn_s_wait_tensorcnt)
#define HAVE_TCNT 1
#endif
#endif

__device__ __forceinline__ float softplusf(float x) {
  return x > 0.f ? x + log1pf(__expf(-x)) : log1pf(__expf(x));
}

// -------------------- utility kernels --------------------

__global__ void zero_f32(float* __restrict__ p, long n) {
  long i = (long)blockIdx.x * blockDim.x + threadIdx.x;
  long s = (long)gridDim.x * blockDim.x;
  for (; i < n; i += s) p[i] = 0.f;
}

// segment-sum over destination nodes: msg[row[e]] += edge_features[e]
__global__ void scatter_add_edges(const float* __restrict__ ef,
                                  const int* __restrict__ row,
                                  float* __restrict__ msg) {
  long t = (long)blockIdx.x * blockDim.x + threadIdx.x;
  if (t >= (long)N_EDGES * 32) return;
  int e = (int)(t >> 5);
  int c = ((int)t & 31) * 4;
  int n = row[e];
  const float4 v = *(const float4*)(ef + (long)e * HDIM + c);
  float* dst = msg + (long)n * HDIM + c;
  atomicAdd(dst + 0, v.x); atomicAdd(dst + 1, v.y);
  atomicAdd(dst + 2, v.z); atomicAdd(dst + 3, v.w);
}

__global__ void cvt2_bf16(const float* __restrict__ a, __bf16* __restrict__ ab,
                          const float* __restrict__ b, __bf16* __restrict__ bb, long n) {
  long i = (long)blockIdx.x * blockDim.x + threadIdx.x;
  long s = (long)gridDim.x * blockDim.x;
  for (; i < n; i += s) { ab[i] = (__bf16)a[i]; bb[i] = (__bf16)b[i]; }
}

// transpose+convert weights to bf16 Bt[f][k] = W[k][f]; fold global feature into cvec
__global__ void prep_weights(const float* __restrict__ Wc, const float* __restrict__ bc,
                             const float* __restrict__ gfeat,
                             const float* __restrict__ W1a, const float* __restrict__ W2a,
                             const float* __restrict__ W1b, const float* __restrict__ W2b,
                             __bf16* __restrict__ WcXt, __bf16* __restrict__ WcMt,
                             __bf16* __restrict__ W1at, __bf16* __restrict__ W2at,
                             __bf16* __restrict__ W1bt, __bf16* __restrict__ W2bt,
                             float* __restrict__ cvec) {
  int i = blockIdx.x * blockDim.x + threadIdx.x;
  const int total = 6 * HDIM * HDIM;
  if (i < total) {
    int mat = i / (HDIM * HDIM);
    int f = (i % (HDIM * HDIM)) / HDIM;  // output row (column of W)
    int k = i % HDIM;                    // output col (row of W)
    float v; __bf16* dst;
    switch (mat) {
      case 0: v = Wc[(long)k * HDIM + f];          dst = WcXt; break;
      case 1: v = Wc[(long)(128 + k) * HDIM + f];  dst = WcMt; break;
      case 2: v = W1a[(long)k * HDIM + f];         dst = W1at; break;
      case 3: v = W2a[(long)k * HDIM + f];         dst = W2at; break;
      case 4: v = W1b[(long)k * HDIM + f];         dst = W1bt; break;
      default: v = W2b[(long)k * HDIM + f];        dst = W2bt; break;
    }
    dst[f * HDIM + k] = (__bf16)v;
  } else if (i < total + HDIM) {
    int f = i - total;
    float acc = bc[f];
    for (int d = 0; d < GDIMF; d++) acc += gfeat[d] * Wc[(long)(256 + d) * HDIM + f];
    cvec[f] = acc;  // bias + global-feature contribution (constant per column)
  }
}

// -------------------- TDM: 2D tile (16 rows x 128 bf16) global -> LDS --------------------
#ifdef HAVE_TDM
__device__ __forceinline__ void tdm_load_tile_2d(unsigned lds_off, const __bf16* gptr) {
  unsigned long long ga = (unsigned long long)gptr;
  // D# group0 (ISA 8.3): count=1 | lds_addr | global_addr[56:0] | type=2
  u32x4 g0;
  g0[0] = 1u;                                   // count=1, user mode, no gather
  g0[1] = lds_off;                              // lds_addr (bytes)
  g0[2] = (unsigned)(ga & 0xffffffffu);         // global_addr[31:0]
  g0[3] = (unsigned)((ga >> 32) & 0x1ffffffu) | (2u << 30);  // global_addr[56:32] | type=2
  // D# group1 (ISA 8.4): wg_mask=0, data_size=1 (2B), dims/strides
  i32x8 g1;
  g1[0] = (1 << 16);        // data_size = 2 bytes
  g1[1] = (HDIM << 16);     // [63:48] tensor_dim0 = 128
  g1[2] = (16 << 16);       // [95:80] tensor_dim1 = 16
  g1[3] = (HDIM << 16);     // [127:112] tile_dim0 = 128
  g1[4] = 16;               // [143:128] tile_dim1 = 16, tile_dim2 = 0
  g1[5] = HDIM;             // tensor_dim0_stride[31:0] = 128 elements
  g1[6] = 0;                // stride0[47:32] = 0, stride1[15:0] = 0
  g1[7] = 0;                // stride1[47:16] = 0
  i32x4 z4 = {0, 0, 0, 0};            // groups 2/3 unused (2D tile)
  i32x8 z8 = {0, 0, 0, 0, 0, 0, 0, 0};
  // amdgpu-toolchain (clang-23) 6-arg form: (g0, g1, g2, g3, g4, cpol)
  __builtin_amdgcn_tensor_load_to_lds(g0, g1, z4, z4, z8, 0);
}
#endif

__device__ __forceinline__ void wait_tensorcnt0() {
#ifdef HAVE_TCNT
  __builtin_amdgcn_s_wait_tensorcnt(0);
#else
  asm volatile("s_wait_tensorcnt 0x0" ::: "memory");
#endif
}

// -------------------- WMMA GEMM --------------------
// C[n,f] = act( A1[n,:]·Bt1[f,:] (+ A2[n,:]·Bt2[f,:]) + bias[f] ), K=128 per operand.
// Block = 256 threads (8 wave32) = one 16-row panel; wave w owns cols [16w,16w+16).
// A tile staged once in LDS (TDM when available); B read from L2-resident weights.

__device__ __forceinline__ v8f wmma_k128(const __bf16* arow,
                                         const __bf16* __restrict__ brow,
                                         int koff, int kb, v8f acc) {
#pragma unroll
  for (int kt = 0; kt < 4; kt++) {
    // A 16x32 bf16 frag (ISA 7.12.2): lane m -> K[koff..koff+8) and K[16+koff..)
    v8bf a0 = *(const v8bf*)(arow + kt * 32 + koff);
    v8bf a1 = *(const v8bf*)(arow + kt * 32 + 16 + koff);
    // B 32x16 frag: lane n -> K[koff..koff+8) and K[16+koff..) for its column
    v8bf b0 = *(const v8bf*)(brow + kt * 32 + koff);
    v8bf b1 = *(const v8bf*)(brow + kt * 32 + 16 + koff);
    v16bf a, b;
#pragma unroll
    for (int i = 0; i < 8; i++) { a[i] = a0[i]; a[i + 8] = a1[i]; b[i] = b0[i]; b[i + 8] = b1[i]; }
    acc = __builtin_amdgcn_wmma_f32_16x16x32_bf16(false, a, false, b, (short)0, acc, false, false);
  }
  return acc;
}

__global__ void wmma_gemm_128(const __bf16* __restrict__ A1, const __bf16* __restrict__ Bt1,
                              const __bf16* __restrict__ A2, const __bf16* __restrict__ Bt2,
                              const float* __restrict__ bias,
                              float* __restrict__ outF, __bf16* __restrict__ outB,
                              int nrows, int act) {
  __shared__ __align__(16) __bf16 Alds[2][16 * HDIM];
  const int lane  = threadIdx.x & 31;
  const int wave  = threadIdx.x >> 5;
  const int mbase = blockIdx.x * 16;
  const int col   = wave * 16 + (lane & 15);
  const int koff  = (lane >> 4) * 8;
  const int kb    = (lane >> 4) * 16;

#ifdef HAVE_TDM
  if (threadIdx.x == 0) {
    // Low 32 bits of the flat shared-memory address are the LDS byte offset (ISA 10.2).
    unsigned lds0 = (unsigned)(unsigned long long)&Alds[0][0];
    tdm_load_tile_2d(lds0, A1 + (size_t)mbase * HDIM);
    if (A2) tdm_load_tile_2d(lds0 + 16 * HDIM * 2, A2 + (size_t)mbase * HDIM);
  }
  wait_tensorcnt0();
#else
  {
    const v8bf* s1 = (const v8bf*)(A1 + (size_t)mbase * HDIM);
    v8bf* d1 = (v8bf*)&Alds[0][0];
    for (int i = threadIdx.x; i < 16 * HDIM / 8; i += blockDim.x) d1[i] = s1[i];
    if (A2) {
      const v8bf* s2 = (const v8bf*)(A2 + (size_t)mbase * HDIM);
      v8bf* d2 = (v8bf*)&Alds[1][0];
      for (int i = threadIdx.x; i < 16 * HDIM / 8; i += blockDim.x) d2[i] = s2[i];
    }
  }
#endif
  __syncthreads();

  v8f acc = {};
  acc = wmma_k128(&Alds[0][(lane & 15) * HDIM], Bt1 + (size_t)col * HDIM, koff, kb, acc);
  if (A2) acc = wmma_k128(&Alds[1][(lane & 15) * HDIM], Bt2 + (size_t)col * HDIM, koff, kb, acc);

  const float bv = bias ? bias[col] : 0.f;
  const int rbase = mbase + ((lane >> 4) * 8);  // D layout: VGPR i -> M = i + 8*(lane>=16)
#pragma unroll
  for (int i = 0; i < 8; i++) {
    int r = rbase + i;
    if (r >= nrows) continue;
    float v = acc[i] + bv;
    if (act) v = softplusf(v);
    if (outF) outF[(size_t)r * HDIM + col] = v;
    if (outB) outB[(size_t)r * HDIM + col] = (__bf16)v;
  }
}

// -------------------- DiffGroupNorm pieces --------------------

__global__ void softmax_groups(const float* __restrict__ h, const float* __restrict__ linW,
                               const float* __restrict__ linb, float* __restrict__ s) {
  int n = blockIdx.x * blockDim.x + threadIdx.x;
  if (n >= N_NODES) return;
  float lg[NGRP];
#pragma unroll
  for (int g = 0; g < NGRP; g++) lg[g] = linb[g];
  const float* hr = h + (size_t)n * HDIM;
  for (int f = 0; f < HDIM; f++) {
    float hv = hr[f];
#pragma unroll
    for (int g = 0; g < NGRP; g++) lg[g] += hv * linW[f * NGRP + g];
  }
  float mx = lg[0];
#pragma unroll
  for (int g = 1; g < NGRP; g++) mx = fmaxf(mx, lg[g]);
  float sum = 0.f;
#pragma unroll
  for (int g = 0; g < NGRP; g++) { lg[g] = __expf(lg[g] - mx); sum += lg[g]; }
  float inv = 1.f / sum;
#pragma unroll
  for (int g = 0; g < NGRP; g++) s[(size_t)n * NGRP + g] = lg[g] * inv;
}

// S1[g,f] = sum_n s[n,g]*h[n,f]; S2[g,f] = sum_n (s*h)^2  (column stats of o)
__global__ void group_stats(const float* __restrict__ h, const float* __restrict__ s,
                            float* __restrict__ S1, float* __restrict__ S2) {
  int f = threadIdx.x;  // 128 threads
  int nper = (N_NODES + gridDim.x - 1) / gridDim.x;
  int n0 = blockIdx.x * nper;
  int n1 = n0 + nper; if (n1 > N_NODES) n1 = N_NODES;
  float a1[NGRP], a2[NGRP];
#pragma unroll
  for (int g = 0; g < NGRP; g++) { a1[g] = 0.f; a2[g] = 0.f; }
  for (int n = n0; n < n1; n++) {
    float hv = h[(size_t)n * HDIM + f];
#pragma unroll
    for (int g = 0; g < NGRP; g++) {
      float p = s[(size_t)n * NGRP + g] * hv;
      a1[g] += p; a2[g] += p * p;
    }
  }
#pragma unroll
  for (int g = 0; g < NGRP; g++) {
    atomicAdd(&S1[g * HDIM + f], a1[g]);
    atomicAdd(&S2[g * HDIM + f], a2[g]);
  }
}

// A[g,f] = rsqrt(var+eps)*gamma; c[f] = sum_g (beta - mu*A)
__global__ void gn_params(const float* __restrict__ S1, const float* __restrict__ S2,
                          const float* __restrict__ gamma, const float* __restrict__ beta,
                          float* __restrict__ Amat, float* __restrict__ cconst) {
  int f = threadIdx.x;  // one block of 128
  const float invN = 1.f / (float)N_NODES;
  float cc = 0.f;
  for (int g = 0; g < NGRP; g++) {
    float mu  = S1[g * HDIM + f] * invN;
    float var = S2[g * HDIM + f] * invN - mu * mu;
    float a   = rsqrtf(var + EPSV) * gamma[g * HDIM + f];
    Amat[g * HDIM + f] = a;
    cc += beta[g * HDIM + f] - mu * a;
  }
  cconst[f] = cc;
}

// new_cur = curIn + h + LAMDA*(h*sum_g s*A + c)  (+ initial for the last block)
__global__ void gn_apply(const float* __restrict__ h, const float* __restrict__ s,
                         const float* __restrict__ Amat, const float* __restrict__ cconst,
                         const float* __restrict__ curIn, const float* __restrict__ initial,
                         float* __restrict__ curOutF, __bf16* __restrict__ curOutB) {
  __shared__ float Ash[NGRP * HDIM];
  __shared__ float ccsh[HDIM];
  for (int i = threadIdx.x; i < NGRP * HDIM; i += blockDim.x) Ash[i] = Amat[i];
  if (threadIdx.x < HDIM) ccsh[threadIdx.x] = cconst[threadIdx.x];
  __syncthreads();
  int f = threadIdx.x & 127;
  int sub = threadIdx.x >> 7;  // two nodes per block-iteration
  for (int n = blockIdx.x * 2 + sub; n < N_NODES; n += gridDim.x * 2) {
    float hv = h[(size_t)n * HDIM + f];
    float dot = 0.f;
#pragma unroll
    for (int g = 0; g < NGRP; g++) dot += s[(size_t)n * NGRP + g] * Ash[g * HDIM + f];
    float nh = hv + LAMDA * (hv * dot + ccsh[f]);
    float c = curIn[(size_t)n * HDIM + f] + nh;
    if (initial) c += initial[(size_t)n * HDIM + f];
    curOutF[(size_t)n * HDIM + f] = c;
    if (curOutB) curOutB[(size_t)n * HDIM + f] = (__bf16)c;
  }
}

// -------------------- host orchestration --------------------

extern "C" void kernel_launch(void* const* d_in, const int* in_sizes, int n_in,
                              void* d_out, int out_size, void* d_ws, size_t ws_size,
                              hipStream_t stream) {
  const float* x    = (const float*)d_in[0];
  const int*   ei   = (const int*)d_in[1];
  const float* ef   = (const float*)d_in[2];
  const float* gfv  = (const float*)d_in[3];
  const float* Wc   = (const float*)d_in[4];
  const float* bc   = (const float*)d_in[5];
  const float* W1a  = (const float*)d_in[6];
  const float* b1a  = (const float*)d_in[7];
  const float* W2a  = (const float*)d_in[8];
  const float* b2a  = (const float*)d_in[9];
  const float* linWa= (const float*)d_in[10];
  const float* linba= (const float*)d_in[11];
  const float* gama = (const float*)d_in[12];
  const float* beta = (const float*)d_in[13];
  const float* W1b  = (const float*)d_in[14];
  const float* b1b  = (const float*)d_in[15];
  const float* W2b  = (const float*)d_in[16];
  const float* b2b  = (const float*)d_in[17];
  const float* linWb= (const float*)d_in[18];
  const float* linbb= (const float*)d_in[19];
  const float* gamb = (const float*)d_in[20];
  const float* betb = (const float*)d_in[21];
  const int* row = ei;  // edge_index[0] = first E entries

  // workspace carve-up (256B aligned)
  char* w = (char*)d_ws;
  auto carve = [&](size_t bytes) -> char* {
    char* p = w; w += (bytes + 255) & ~((size_t)255); return p;
  };
  const size_t NH = (size_t)N_NODES * HDIM;
  float*  msg    = (float*)carve(NH * 4);
  float*  out0   = (float*)carve(NH * 4);
  float*  cur    = (float*)carve(NH * 4);
  float*  hbuf   = (float*)carve(NH * 4);
  __bf16* xbf    = (__bf16*)carve(NH * 2);
  __bf16* msgbf  = (__bf16*)carve(NH * 2);
  __bf16* curbf  = (__bf16*)carve(NH * 2);
  __bf16* tbf    = (__bf16*)carve(NH * 2);
  float*  sprob  = (float*)carve((size_t)N_NODES * NGRP * 4);
  __bf16* WcXt   = (__bf16*)carve(HDIM * HDIM * 2);
  __bf16* WcMt   = (__bf16*)carve(HDIM * HDIM * 2);
  __bf16* W1at   = (__bf16*)carve(HDIM * HDIM * 2);
  __bf16* W2at   = (__bf16*)carve(HDIM * HDIM * 2);
  __bf16* W1bt   = (__bf16*)carve(HDIM * HDIM * 2);
  __bf16* W2bt   = (__bf16*)carve(HDIM * HDIM * 2);
  float*  cvec   = (float*)carve(HDIM * 4);
  float*  stats  = (float*)carve(4 * NGRP * HDIM * 4);  // S1a,S2a,S1b,S2b contiguous
  float*  S1a = stats, *S2a = stats + NGRP * HDIM;
  float*  S1b = stats + 2 * NGRP * HDIM, *S2b = stats + 3 * NGRP * HDIM;
  float*  Aa   = (float*)carve(NGRP * HDIM * 4);
  float*  cca  = (float*)carve(HDIM * 4);
  float*  Ab   = (float*)carve(NGRP * HDIM * 4);
  float*  ccb  = (float*)carve(HDIM * 4);

  const int TPB = 256;
  // 1) zero scatter target + stats accumulators
  zero_f32<<<2048, TPB, 0, stream>>>(msg, (long)NH);
  zero_f32<<<20, TPB, 0, stream>>>(stats, (long)4 * NGRP * HDIM);
  // 2) segment-sum of edge features (f32 atomics, target L2-resident)
  {
    long tot = (long)N_EDGES * 32;
    scatter_add_edges<<<(unsigned)((tot + TPB - 1) / TPB), TPB, 0, stream>>>(ef, row, msg);
  }
  // 3) weight transpose/convert + folded global-feature bias
  {
    int tot = 6 * HDIM * HDIM + HDIM;
    prep_weights<<<(tot + TPB - 1) / TPB, TPB, 0, stream>>>(
        Wc, bc, gfv, W1a, W2a, W1b, W2b, WcXt, WcMt, W1at, W2at, W1bt, W2bt, cvec);
  }
  // 4) bf16 copies of x and messages (GEMM operands)
  cvt2_bf16<<<2048, TPB, 0, stream>>>(x, xbf, msg, msgbf, (long)NH);

  const int GB = N_NODES / 16;  // 3125 row-panels of 16
  // 5) combine GEMM: out0 = softplus(x@WcX + msg@WcM + cvec); also bf16 copy as block-a input
  wmma_gemm_128<<<GB, TPB, 0, stream>>>(xbf, WcXt, msgbf, WcMt, cvec, out0, curbf, N_NODES, 1);

  // ---- block a ----
  wmma_gemm_128<<<GB, TPB, 0, stream>>>(curbf, W1at, nullptr, nullptr, b1a, nullptr, tbf, N_NODES, 1);
  wmma_gemm_128<<<GB, TPB, 0, stream>>>(tbf, W2at, nullptr, nullptr, b2a, hbuf, nullptr, N_NODES, 0);
  softmax_groups<<<(N_NODES + TPB - 1) / TPB, TPB, 0, stream>>>(hbuf, linWa, linba, sprob);
  group_stats<<<512, 128, 0, stream>>>(hbuf, sprob, S1a, S2a);
  gn_params<<<1, 128, 0, stream>>>(S1a, S2a, gama, beta, Aa, cca);
  gn_apply<<<1024, TPB, 0, stream>>>(hbuf, sprob, Aa, cca, out0, nullptr, cur, curbf);

  // ---- block b ----
  wmma_gemm_128<<<GB, TPB, 0, stream>>>(curbf, W1bt, nullptr, nullptr, b1b, nullptr, tbf, N_NODES, 1);
  wmma_gemm_128<<<GB, TPB, 0, stream>>>(tbf, W2bt, nullptr, nullptr, b2b, hbuf, nullptr, N_NODES, 0);
  softmax_groups<<<(N_NODES + TPB - 1) / TPB, TPB, 0, stream>>>(hbuf, linWb, linbb, sprob);
  group_stats<<<512, 128, 0, stream>>>(hbuf, sprob, S1b, S2b);
  gn_params<<<1, 128, 0, stream>>>(S1b, S2b, gamb, betb, Ab, ccb);
  // final: d_out = cur + h_b_norm + initial
  gn_apply<<<1024, TPB, 0, stream>>>(hbuf, sprob, Ab, ccb, cur, out0, (float*)d_out, nullptr);

  (void)in_sizes; (void)n_in; (void)out_size; (void)ws_size;
}
